// VariationalAttention_850403525219
// MI455X (gfx1250) — matrix operand order repaired and checked by reading, not compiled, not run on hardware
//
#include <hip/hip_runtime.h>
#include <hip/hip_bf16.h>

// ---------------------------------------------------------------------------
// VariationalAttention on gfx1250 (MI455X, wave32, WMMA, bf16 operand path)
// N=16, T=64, S=1024, SRC=TGT=512, D=64
//
// Data path: one-time f32->bf16 conversion passes (activations row-major,
// weights transposed to [N,K]) so every WMMA fragment is a contiguous
// b128-vector load; 4 N-tiles per wave reuse the A fragment across 4 WMMAs.
// Transcendental stage uses the CDNA5 hardware v_tanh_f32 TRANS op, batched
// 4-wide so consecutive tanh ops satisfy the TRANS RAW hazard for each other.
// ---------------------------------------------------------------------------

typedef __bf16 bf16_t;
typedef __attribute__((ext_vector_type(16))) __bf16 v16bf;
typedef __attribute__((ext_vector_type(8)))  __bf16 v8bf;
typedef __attribute__((ext_vector_type(8)))  float  v8f;

#define Nb   16
#define Tt   64
#define Ss   1024
#define Dd   64
#define SRCd 512
#define TGTd 512

// round-to-nearest-even f32 -> bf16 (integer path)
__device__ __forceinline__ bf16_t f2bf(float f) {
  unsigned u = __builtin_bit_cast(unsigned, f);
  unsigned r = (u + 0x7FFFu + ((u >> 16) & 1u)) >> 16;
  return __builtin_bit_cast(bf16_t, (unsigned short)r);
}

// CDNA5 hardware tanh (TRANS32).
#if __has_builtin(__builtin_amdgcn_tanhf)
__device__ __forceinline__ float htanh(float x) { return __builtin_amdgcn_tanhf(x); }
__device__ __forceinline__ void htanh4(float& r0, float& r1, float& r2, float& r3,
                                       float x0, float x1, float x2, float x3) {
  r0 = __builtin_amdgcn_tanhf(x0);
  r1 = __builtin_amdgcn_tanhf(x1);
  r2 = __builtin_amdgcn_tanhf(x2);
  r3 = __builtin_amdgcn_tanhf(x3);
}
#else
// v_nop covers the TRANS RAW hazard (ISA 7.4) after the last tanh; inside the
// 4-wide block each tanh is the required independent op for its predecessor.
__device__ __forceinline__ float htanh(float x) {
  float r;
  asm volatile("v_tanh_f32 %0, %1\n\tv_nop" : "=v"(r) : "v"(x));
  return r;
}
__device__ __forceinline__ void htanh4(float& r0, float& r1, float& r2, float& r3,
                                       float x0, float x1, float x2, float x3) {
  asm volatile("v_tanh_f32 %0, %4\n\t"
               "v_tanh_f32 %1, %5\n\t"
               "v_tanh_f32 %2, %6\n\t"
               "v_tanh_f32 %3, %7\n\t"
               "v_nop"
               : "=&v"(r0), "=&v"(r1), "=&v"(r2), "=&v"(r3)
               : "v"(x0), "v"(x1), "v"(x2), "v"(x3));
}
#endif

// ---------------------------------------------------------------------------
// fragment loads (contiguous k order used by this code for both A and B, not the ISA table order)
//   A 16x32 bf16: lane row = lane&15, h = lane>>4,
//     elems 0..15 -> k = k0 + 16h + e       (32 contiguous bytes)
//   B 32x16 bf16 (stored transposed [N,K]): lane col = lane&15,
//     elems 0..15 -> k = k0 + 16h + e       (32 contiguous bytes)
// ---------------------------------------------------------------------------
__device__ __forceinline__ v16bf ldA(const bf16_t* __restrict__ A, int lda,
                                     int row, int k0, int h) {
  const bf16_t* p = A + (long)row * lda + k0 + 16 * h;
  v8bf lo = *(const v8bf*)(p);
  v8bf hi = *(const v8bf*)(p + 8);
  return __builtin_shufflevector(lo, hi, 0, 1, 2, 3, 4, 5, 6, 7,
                                 8, 9, 10, 11, 12, 13, 14, 15);
}

__device__ __forceinline__ v16bf ldB(const bf16_t* __restrict__ Bt, int ldb,
                                     int col, int k0, int h) {
  return *(const v16bf*)(Bt + (long)col * ldb + k0 + 16 * h);
}

// 16x64 output per wave: A fragment reused across 4 WMMAs
__device__ __forceinline__ void gemm_acc4(const bf16_t* __restrict__ A,
                                          const bf16_t* __restrict__ Bt,
                                          int K, int lda, int ldb,
                                          int m0, int n0, int lane, v8f acc[4]) {
  const int h = lane >> 4, mr = lane & 15;
  for (int k0 = 0; k0 < K; k0 += 32) {
    v16bf a = ldA(A, lda, m0 + mr, k0, h);
#pragma unroll
    for (int t = 0; t < 4; ++t) {
      v16bf b = ldB(Bt, ldb, n0 + 16 * t + mr, k0, h);
      acc[t] = __builtin_amdgcn_wmma_f32_16x16x32_bf16(false, a, false, b,
                                                       (short)0, acc[t],
                                                       false, false);
    }
  }
}

// ---------------------------------------------------------------------------
// GEMM kernels: grid = (Nout/64, M/16, batch), block = 32 (one wave)
// ---------------------------------------------------------------------------
__global__ void gemm_f32out(const bf16_t* __restrict__ A,
                            const bf16_t* __restrict__ Bt,
                            float* __restrict__ C,
                            int K, int lda, int ldb, int ldc,
                            long sA, long sB, long sC) {
  const int n0 = blockIdx.x * 64, m0 = blockIdx.y * 16;
  const long b = blockIdx.z;
  A += b * sA; Bt += b * sB; C += b * sC;
  const int lane = threadIdx.x & 31, h = lane >> 4, mr = lane & 15;
  v8f acc[4] = {};
  gemm_acc4(A, Bt, K, lda, ldb, m0, n0, lane, acc);
#pragma unroll
  for (int t = 0; t < 4; ++t)
#pragma unroll
    for (int e = 0; e < 8; ++e)
      C[(long)(m0 + e + 8 * h) * ldc + n0 + 16 * t + mr] = acc[t][e];
}

__global__ void gemm_bf16out(const bf16_t* __restrict__ A,
                             const bf16_t* __restrict__ Bt,
                             bf16_t* __restrict__ C,
                             int K, int lda, int ldb, int ldc,
                             long sA, long sB, long sC) {
  const int n0 = blockIdx.x * 64, m0 = blockIdx.y * 16;
  const long b = blockIdx.z;
  A += b * sA; Bt += b * sB; C += b * sC;
  const int lane = threadIdx.x & 31, h = lane >> 4, mr = lane & 15;
  v8f acc[4] = {};
  gemm_acc4(A, Bt, K, lda, ldb, m0, n0, lane, acc);
#pragma unroll
  for (int t = 0; t < 4; ++t)
#pragma unroll
    for (int e = 0; e < 8; ++e)
      C[(long)(m0 + e + 8 * h) * ldc + n0 + 16 * t + mr] = f2bf(acc[t][e]);
}

// out = tanh(A*Bt + iw + bias), f32 store
__global__ void gemm_tanh_out(const bf16_t* __restrict__ A,
                              const bf16_t* __restrict__ Bt,
                              const float* __restrict__ iw,
                              const float* __restrict__ bias,
                              float* __restrict__ C,
                              int K, int lda, int ldb, int ldc) {
  const int n0 = blockIdx.x * 64, m0 = blockIdx.y * 16;
  const int lane = threadIdx.x & 31, h = lane >> 4, mr = lane & 15;
  v8f acc[4] = {};
  gemm_acc4(A, Bt, K, lda, ldb, m0, n0, lane, acc);
#pragma unroll
  for (int t = 0; t < 4; ++t)
#pragma unroll
    for (int e = 0; e < 8; ++e) {
      const long m = m0 + e + 8 * h;
      const int n = n0 + 16 * t + mr;
      C[m * ldc + n] = htanh(acc[t][e] + iw[m * ldc + n] + bias[n]);
    }
}

// ---------------------------------------------------------------------------
// conversion passes
// ---------------------------------------------------------------------------
// flat f32 -> bf16 (row-major preserved); 4 elements per thread
__global__ __launch_bounds__(256) void cvt_bf16(const float* __restrict__ in,
                                                bf16_t* __restrict__ out,
                                                long nElem) {
  const long i = ((long)blockIdx.x * 256 + threadIdx.x) * 4;
  if (i + 3 < nElem) {
    const float4 f = *(const float4*)(in + i);
    ushort2 lo, hi;
    lo.x = __builtin_bit_cast(unsigned short, f2bf(f.x));
    lo.y = __builtin_bit_cast(unsigned short, f2bf(f.y));
    hi.x = __builtin_bit_cast(unsigned short, f2bf(f.z));
    hi.y = __builtin_bit_cast(unsigned short, f2bf(f.w));
    *(ushort2*)(out + i) = lo;
    *(ushort2*)(out + i + 2) = hi;
  }
}

// [R,C] f32 -> [C,R] bf16, LDS-tiled; grid (C/32, R/32, batch), block (32,8)
__global__ __launch_bounds__(256) void transpose_cvt(const float* __restrict__ in,
                                                     bf16_t* __restrict__ out,
                                                     int R, int C,
                                                     long inStride, long outStride) {
  __shared__ float tile[32][33];
  const int c0 = blockIdx.x * 32, r0 = blockIdx.y * 32;
  const long b = blockIdx.z;
  in += b * inStride; out += b * outStride;
  const int tx = threadIdx.x, ty = threadIdx.y;
#pragma unroll
  for (int j = 0; j < 4; ++j)
    tile[ty + j * 8][tx] = in[(long)(r0 + ty + j * 8) * C + (c0 + tx)];
  __syncthreads();
#pragma unroll
  for (int j = 0; j < 4; ++j)
    out[(long)(c0 + ty + j * 8) * R + (r0 + tx)] = f2bf(tile[tx][ty + j * 8]);
}

// ---------------------------------------------------------------------------
// JAX threefry2x32 (key = PRNGKey(42) = {0,42}), iota-counter split-half scheme
// ---------------------------------------------------------------------------
__device__ __forceinline__ unsigned rotl32(unsigned x, int r) {
  return (x << r) | (x >> (32 - r));
}

__device__ __forceinline__ unsigned threefry_bits(unsigned i) {
  const unsigned half = 1u << 19;                 // N*T*S/2 = 2^19
  const bool first = i < half;
  unsigned x0 = first ? i : i - half;
  unsigned x1 = first ? i + half : i;
  const unsigned ks0 = 0u, ks1 = 42u, ks2 = 0x1BD11BDAu ^ ks0 ^ ks1;
  x0 += ks0; x1 += ks1;
  const int rotA[4] = {13, 15, 26, 6};
  const int rotB[4] = {17, 29, 16, 24};
#define R4(RR) { _Pragma("unroll") for (int j = 0; j < 4; ++j) { x0 += x1; x1 = rotl32(x1, RR[j]); x1 ^= x0; } }
  R4(rotA); x0 += ks1; x1 += ks2 + 1u;
  R4(rotB); x0 += ks2; x1 += ks0 + 2u;
  R4(rotA); x0 += ks0; x1 += ks1 + 3u;
  R4(rotB); x0 += ks1; x1 += ks2 + 4u;
  R4(rotA); x0 += ks2; x1 += ks0 + 5u;
#undef R4
  return first ? x0 : x1;
}

// ---------------------------------------------------------------------------
// Scores + log_softmax + alpha + gumbel-softmax y (bf16 outputs)
// scores[s] = sum_d v[d] * tanh(wq[n,t,d] + uh[n,s,d])
// ---------------------------------------------------------------------------
__global__ __launch_bounds__(256) void scores_gumbel(
    const float* __restrict__ wq, const float* __restrict__ uh,
    const float* __restrict__ v,
    bf16_t* __restrict__ alpha, bf16_t* __restrict__ yal) {
  const int nt  = blockIdx.x;          // 0 .. N*T-1
  const int n   = nt >> 6;             // T = 64
  const int tid = threadIdx.x;

  __shared__ float q[Dd], vv[Dd];
  __shared__ float sc[Ss];
  __shared__ float red[256];

  if (tid < Dd) { q[tid] = wq[nt * Dd + tid]; vv[tid] = v[tid]; }
  __syncthreads();

  const float* uhn = uh + (long)n * Ss * Dd;
  for (int s = tid; s < Ss; s += 256) {
    const float* row = uhn + s * Dd;
    if (s + 256 < Ss)
      __builtin_prefetch(uhn + (s + 256) * Dd, 0, 0);  // global_prefetch_b8
    float acc = 0.f;
#pragma unroll
    for (int d = 0; d < Dd; d += 4) {
      float t0, t1, t2, t3;
      htanh4(t0, t1, t2, t3,
             q[d] + row[d], q[d + 1] + row[d + 1],
             q[d + 2] + row[d + 2], q[d + 3] + row[d + 3]);
      acc += vv[d] * t0 + vv[d + 1] * t1 + vv[d + 2] * t2 + vv[d + 3] * t3;
    }
    sc[s] = acc;
  }
  __syncthreads();

  float lm = -1e30f;
  for (int s = tid; s < Ss; s += 256) lm = fmaxf(lm, sc[s]);
  red[tid] = lm; __syncthreads();
  for (int off = 128; off > 0; off >>= 1) {
    if (tid < off) red[tid] = fmaxf(red[tid], red[tid + off]);
    __syncthreads();
  }
  const float mx = red[0]; __syncthreads();

  float ls = 0.f;
  for (int s = tid; s < Ss; s += 256) ls += __expf(sc[s] - mx);
  red[tid] = ls; __syncthreads();
  for (int off = 128; off > 0; off >>= 1) {
    if (tid < off) red[tid] += red[tid + off];
    __syncthreads();
  }
  const float lse = __logf(red[0]) + mx; __syncthreads();

  float zloc[4];
#pragma unroll
  for (int i = 0; i < 4; ++i) {
    const int s = tid + i * 256;
    const float la = sc[s] - lse;
    alpha[(long)nt * Ss + s] = f2bf(__expf(la));
    const unsigned bits = threefry_bits((unsigned)nt * (unsigned)Ss + (unsigned)s);
    const float u = __builtin_bit_cast(float, (bits >> 9) | 0x3f800000u) - 1.0f;
    const float g = -__logf(-__logf(u + 1e-20f) + 1e-20f);
    zloc[i] = (la + g) * 2.0f;   // /TEMPERATURE
  }
  __syncthreads();
#pragma unroll
  for (int i = 0; i < 4; ++i) sc[tid + i * 256] = zloc[i];
  __syncthreads();

  lm = -1e30f;
  for (int s = tid; s < Ss; s += 256) lm = fmaxf(lm, sc[s]);
  red[tid] = lm; __syncthreads();
  for (int off = 128; off > 0; off >>= 1) {
    if (tid < off) red[tid] = fmaxf(red[tid], red[tid + off]);
    __syncthreads();
  }
  const float mx2 = red[0]; __syncthreads();
  ls = 0.f;
  for (int s = tid; s < Ss; s += 256) ls += __expf(sc[s] - mx2);
  red[tid] = ls; __syncthreads();
  for (int off = 128; off > 0; off >>= 1) {
    if (tid < off) red[tid] += red[tid + off];
    __syncthreads();
  }
  const float inv = 1.0f / red[0]; __syncthreads();
  for (int s = tid; s < Ss; s += 256)
    yal[(long)nt * Ss + s] = f2bf(__expf(sc[s] - mx2) * inv);
}

// ---------------------------------------------------------------------------
extern "C" void kernel_launch(void* const* d_in, const int* in_sizes, int n_in,
                              void* d_out, int out_size, void* d_ws, size_t ws_size,
                              hipStream_t stream) {
  const float* input   = (const float*)d_in[0];   // [16,64,512]
  const float* membank = (const float*)d_in[1];   // [16,1024,512]
  const float* W_q     = (const float*)d_in[2];   // [512,64]
  const float* W_ctx   = (const float*)d_in[3];   // [512,64]
  const float* v       = (const float*)d_in[4];   // [64]
  const float* W_out   = (const float*)d_in[5];   // [1024,512]
  const float* b_out   = (const float*)d_in[6];   // [512]
  float* out = (float*)d_out;                     // [2,16,64,512]
  (void)ws_size; (void)in_sizes; (void)n_in; (void)out_size;

  // workspace layout (256B aligned chunks)
  char* p = (char*)d_ws;
  auto alloc = [&](size_t bytes) { char* r = p; p += (bytes + 255) & ~(size_t)255; return r; };
  bf16_t* inputbf   = (bf16_t*)alloc((size_t)Nb * Tt * TGTd * 2);       //  1 MB
  bf16_t* membankbf = (bf16_t*)alloc((size_t)Nb * Ss * SRCd * 2);       // 16.8 MB
  bf16_t* membankT  = (bf16_t*)alloc((size_t)Nb * SRCd * Ss * 2);       // 16.8 MB
  bf16_t* WqT       = (bf16_t*)alloc((size_t)Dd * TGTd * 2);
  bf16_t* WctxT     = (bf16_t*)alloc((size_t)Dd * SRCd * 2);
  bf16_t* WoT1      = (bf16_t*)alloc((size_t)TGTd * TGTd * 2);
  bf16_t* WoT2      = (bf16_t*)alloc((size_t)TGTd * SRCd * 2);
  float*  wq        = (float*) alloc((size_t)Nb * Tt * Dd * 4);
  float*  uh        = (float*) alloc((size_t)Nb * Ss * Dd * 4);
  bf16_t* alphabf   = (bf16_t*)alloc((size_t)Nb * Tt * Ss * 2);
  bf16_t* yalbf     = (bf16_t*)alloc((size_t)Nb * Tt * Ss * 2);
  float*  iw        = (float*) alloc((size_t)Nb * Tt * TGTd * 4);
  bf16_t* ctxc      = (bf16_t*)alloc((size_t)Nb * Tt * SRCd * 2);
  bf16_t* ctxy      = (bf16_t*)alloc((size_t)Nb * Tt * SRCd * 2);

  const dim3 wave(32);
  const dim3 tblk(32, 8);

  // ---- conversion passes ----
  {
    long n1 = (long)Nb * Tt * TGTd;
    cvt_bf16<<<dim3((unsigned)((n1 / 4 + 255) / 256)), 256, 0, stream>>>(input, inputbf, n1);
    long n2 = (long)Nb * Ss * SRCd;
    cvt_bf16<<<dim3((unsigned)((n2 / 4 + 255) / 256)), 256, 0, stream>>>(membank, membankbf, n2);
  }
  transpose_cvt<<<dim3(Dd / 32, TGTd / 32, 1), tblk, 0, stream>>>(W_q, WqT, TGTd, Dd, 0, 0);
  transpose_cvt<<<dim3(Dd / 32, SRCd / 32, 1), tblk, 0, stream>>>(W_ctx, WctxT, SRCd, Dd, 0, 0);
  transpose_cvt<<<dim3(TGTd / 32, TGTd / 32, 1), tblk, 0, stream>>>(W_out, WoT1, TGTd, TGTd, 0, 0);
  transpose_cvt<<<dim3(SRCd / 32, TGTd / 32, 1), tblk, 0, stream>>>(W_out + (long)TGTd * TGTd, WoT2, TGTd, SRCd, 0, 0);
  transpose_cvt<<<dim3(SRCd / 32, Ss / 32, Nb), tblk, 0, stream>>>(
      membank, membankT, Ss, SRCd, (long)Ss * SRCd, (long)SRCd * Ss);

  // ---- projections ----
  // wq = input @ W_q : [1024x512]x[512x64]
  gemm_f32out<<<dim3(1, (Nb * Tt) / 16, 1), wave, 0, stream>>>(
      inputbf, WqT, wq, TGTd, TGTd, TGTd, Dd, 0, 0, 0);
  // uh = memory_bank @ W_ctx : [16384x512]x[512x64]
  gemm_f32out<<<dim3(1, (Nb * Ss) / 16, 1), wave, 0, stream>>>(
      membankbf, WctxT, uh, SRCd, SRCd, SRCd, Dd, 0, 0, 0);
  // iw = input @ W_out[:512,:] : [1024x512]x[512x512]
  gemm_f32out<<<dim3(TGTd / 64, (Nb * Tt) / 16, 1), wave, 0, stream>>>(
      inputbf, WoT1, iw, TGTd, TGTd, TGTd, TGTd, 0, 0, 0);

  // ---- scores + softmax + gumbel ----
  scores_gumbel<<<Nb * Tt, 256, 0, stream>>>(wq, uh, v, alphabf, yalbf);

  // ---- contexts: batched [64x1024]x[1024x512] ----
  gemm_bf16out<<<dim3(SRCd / 64, Tt / 16, Nb), wave, 0, stream>>>(
      alphabf, membankT, ctxc, Ss, Ss, Ss, SRCd,
      (long)Tt * Ss, (long)SRCd * Ss, (long)Tt * SRCd);
  gemm_bf16out<<<dim3(SRCd / 64, Tt / 16, Nb), wave, 0, stream>>>(
      yalbf, membankT, ctxy, Ss, Ss, Ss, SRCd,
      (long)Tt * Ss, (long)SRCd * Ss, (long)Tt * SRCd);

  // ---- outputs: tanh(ctx @ W_out[512:,:] + iw + b) ----
  gemm_tanh_out<<<dim3(TGTd / 64, (Nb * Tt) / 16), wave, 0, stream>>>(
      ctxc, WoT2, iw, b_out, out, SRCd, SRCd, SRCd, TGTd);
  gemm_tanh_out<<<dim3(TGTd / 64, (Nb * Tt) / 16), wave, 0, stream>>>(
      ctxy, WoT2, iw, b_out, out + (long)Nb * Tt * TGTd, SRCd, SRCd, SRCd, TGTd);
}
